// ESABotGCN_32590211842598
// MI455X (gfx1250) — compile-verified, matrix operand order checked by reading, not run on hardware
//
#include <hip/hip_runtime.h>

// ---------------------------------------------------------------------------
// ESABotGCN on MI455X (gfx1250, wave32, WMMA f32_16x16x32_f16)
// Weights pre-packed into per-lane WMMA B-fragment layout (f16) so the hot
// GEMM loops do only contiguous b128 loads + wmma.
// ---------------------------------------------------------------------------

typedef __attribute__((ext_vector_type(16))) _Float16 v16h;
typedef __attribute__((ext_vector_type(8)))  float    v8f;

#define SLOPE 0.01f

__device__ __forceinline__ float lrelu(float x) { return x > 0.f ? x : SLOPE * x; }

__device__ __forceinline__ v8f wmma_f16(v16h a, v16h b, v8f c) {
    // D = A(16x32 f16) * B(32x16 f16) + C(16x16 f32)
    return __builtin_amdgcn_wmma_f32_16x16x32_f16(
        /*neg_a=*/false, a, /*neg_b=*/false, b,
        /*c_mod=*/(short)0, c, /*reuse_a=*/false, /*reuse_b=*/false);
}

// A-fragment: 16x32 tile of row-major f32 matrix X (row stride ld), rows
// [row0,row0+16), K columns [kk,kk+32). ISA layout:
//   lane l<16 : M=l,   K = kk+{0..7}  in h[0..7],  kk+{16..23} in h[8..15]
//   lane l>=16: M=l-16,K = kk+{8..15} in h[0..7],  kk+{24..31} in h[8..15]
__device__ __forceinline__ v16h load_a_frag(const float* X, int row0, int ld,
                                            int kk, int rmax) {
    int lane = threadIdx.x & 31;
    int m    = lane & 15;
    int ksel = (lane >> 4) << 3;          // 0 or 8
    int r = row0 + m; if (r >= rmax) r = rmax - 1;   // clamp (store is guarded)
    const float* p = X + (long)r * ld + kk + ksel;
    v16h a;
#pragma unroll
    for (int j = 0; j < 8; ++j) {
        a[j]     = (_Float16)p[j];
        a[8 + j] = (_Float16)p[16 + j];
    }
    return a;
}

// ---------------------------------------------------------------------------
// Weight pre-pack: W[K x NOUT] f32 row-major -> f16 B fragments, fragment
// fi = nt*KS + t (KS = K/32 k-steps), element [fi*32 + lane] = v16h with the
// K packing symmetric to A; columns >= NOUT zero-padded. One thread = one
// (fragment, lane) pair.
// ---------------------------------------------------------------------------
__global__ void pack_b_kernel(const float* __restrict__ W, v16h* __restrict__ P,
                              int KS, int NOUT, int NT) {
    int tid = blockIdx.x * blockDim.x + threadIdx.x;
    if (tid >= NT * KS * 32) return;
    int lane = tid & 31;
    int fi   = tid >> 5;
    int t    = fi % KS;
    int nt   = fi / KS;
    int n    = nt * 16 + (lane & 15);
    int ksel = (lane >> 4) << 3;
    int kk   = t * 32;
    v16h b;
    if (n < NOUT) {
        const float* p = W + (long)(kk + ksel) * NOUT + n;
#pragma unroll
        for (int j = 0; j < 8; ++j) {
            b[j]     = (_Float16)p[(long)j * NOUT];
            b[8 + j] = (_Float16)p[(long)(16 + j) * NOUT];
        }
    } else {
#pragma unroll
        for (int j = 0; j < 16; ++j) b[j] = (_Float16)0.f;
    }
    P[tid] = b;
}

// GEMM segment (encoder): Y[:, out_off..out_off+NOUT) = act(X[16,K] @ W + b)
template <int K, int NOUT, int NT, bool RELU>
__device__ __forceinline__ void gemm_seg(const float* X, int ldx,
                                         const v16h* __restrict__ Wp,
                                         const float* __restrict__ bias,
                                         float* Y, int ldy, int out_off,
                                         int row0, int rmax) {
    constexpr int KS = K / 32;
    int lane = threadIdx.x & 31;
    v8f c[NT] = {};
    for (int t = 0; t < KS; ++t) {
        v16h a = load_a_frag(X, row0, ldx, t * 32, rmax);
#pragma unroll
        for (int nt = 0; nt < NT; ++nt)
            c[nt] = wmma_f16(a, Wp[(nt * KS + t) * 32 + lane], c[nt]);
    }
    int ncol  = lane & 15;
    int rbase = row0 + ((lane >> 4) << 3);      // D rows: (lane>=16?8:0)+r
#pragma unroll
    for (int nt = 0; nt < NT; ++nt) {
        int col = nt * 16 + ncol;
        if (col >= NOUT) continue;
        float bv = bias[col];
#pragma unroll
        for (int r = 0; r < 8; ++r) {
            int row = rbase + r;
            if (row < rmax) {
                float v = c[nt][r] + bv;
                Y[(long)row * ldy + out_off + col] = RELU ? lrelu(v) : v;
            }
        }
    }
}

// ---------------------------------------------------------------------------
// Kernel 1: fused feature encoder -> x[N,128] = lrelu(concat(...))
// One wave per 16-row node tile; WMMA for the 768-wide encoders, VALU for
// the tiny (K=7/11/1) encoders.
// ---------------------------------------------------------------------------
__global__ __launch_bounds__(256) void encoder_kernel(
    const float* __restrict__ des, const float* __restrict__ tweet,
    const float* __restrict__ num, const float* __restrict__ cat,
    const float* __restrict__ nf,
    const v16h* __restrict__ pWd, const float* __restrict__ bd,
    const v16h* __restrict__ pWt, const float* __restrict__ bt,
    const float* __restrict__ Wn, const float* __restrict__ bn,
    const float* __restrict__ Wc, const float* __restrict__ bc,
    const float* __restrict__ Wf, const float* __restrict__ bf,
    float* __restrict__ X, int n) {
    int wave = blockIdx.x * (blockDim.x >> 5) + (threadIdx.x >> 5);
    int row0 = wave * 16;
    if (row0 >= n) return;

    gemm_seg<768, 28, 2, true>(des,   768, pWd, bd, X, 128,  0, row0, n);
    gemm_seg<768, 36, 3, true>(tweet, 768, pWt, bt, X, 128, 28, row0, n);

    // cols 64..75 = num(7->12), 76..115 = cat(11->40), 116..127 = nf(1->12)
    int lane = threadIdx.x & 31;
    for (int idx = lane; idx < 16 * 64; idx += 32) {
        int row = row0 + (idx >> 6);
        if (row >= n) continue;
        int c = idx & 63;
        float acc;
        if (c < 12) {
            acc = bn[c];
#pragma unroll
            for (int k = 0; k < 7; ++k) acc += num[(long)row * 7 + k] * Wn[k * 12 + c];
        } else if (c < 52) {
            int cc = c - 12;
            acc = bc[cc];
#pragma unroll
            for (int k = 0; k < 11; ++k) acc += cat[(long)row * 11 + k] * Wc[k * 40 + cc];
        } else {
            int cc = c - 52;
            acc = bf[cc] + nf[row] * Wf[cc];
        }
        X[(long)row * 128 + 64 + c] = lrelu(acc);
    }
}

// ---------------------------------------------------------------------------
// Kernel 2: generic Y[N,NOUT] = act(X[N,128] @ W[128,NOUT] (+ b))
// In-place safe when X==Y: all A fragments are loaded before any store and
// each wave touches only its own 16 rows (no __restrict__ on X/Y).
// ---------------------------------------------------------------------------
template <int NOUT, bool RELU, bool BIAS>
__global__ __launch_bounds__(256) void gemm128_kernel(
    const float* X, const v16h* __restrict__ Wp,
    const float* __restrict__ bias, float* Y, int n) {
    int wave = blockIdx.x * (blockDim.x >> 5) + (threadIdx.x >> 5);
    int row0 = wave * 16;
    if (row0 >= n) return;

    v16h a[4];
#pragma unroll
    for (int t = 0; t < 4; ++t) a[t] = load_a_frag(X, row0, 128, t * 32, n);

    constexpr int NT = (NOUT + 15) / 16;
    int lane  = threadIdx.x & 31;
    int ncol  = lane & 15;
    int rbase = row0 + ((lane >> 4) << 3);
#pragma unroll
    for (int nt = 0; nt < NT; ++nt) {
        v8f c = {};
#pragma unroll
        for (int t = 0; t < 4; ++t)
            c = wmma_f16(a[t], Wp[(nt * 4 + t) * 32 + lane], c);
        int col = nt * 16 + ncol;
        if (col < NOUT) {
            float bv = BIAS ? bias[col] : 0.f;
#pragma unroll
            for (int r = 0; r < 8; ++r) {
                int row = rbase + r;
                if (row < n) {
                    float v = c[r] + bv;
                    Y[(long)row * NOUT + col] = RELU ? lrelu(v) : v;
                }
            }
        }
    }
}

// ---------------------------------------------------------------------------
// Graph-normalization + scatter kernels
// ---------------------------------------------------------------------------
__global__ void fill_kernel(float* p, float v, long n) {
    long i = (long)blockIdx.x * blockDim.x + threadIdx.x;
    if (i < n) p[i] = v;
}

__global__ void deg_kernel(float* deg, const int* __restrict__ dst, int E) {
    long e = (long)blockIdx.x * blockDim.x + threadIdx.x;
    if (e < E) atomicAdd(&deg[dst[e]], 1.0f);
}

__global__ void rsqrt_kernel(float* d, int n) {
    long i = (long)blockIdx.x * blockDim.x + threadIdx.x;
    if (i < n) d[i] = rsqrtf(d[i]);   // deg >= 1 (self loop) -> never div0
}

// agg[dst] += h[src] * dis[src]*dis[dst] ; one edge per 32 lanes (float4/lane)
__global__ __launch_bounds__(256) void scatter_kernel(
    const float* __restrict__ h, float* agg, const float* __restrict__ dis,
    const int* __restrict__ src, const int* __restrict__ dst, int E) {
    long t = (long)blockIdx.x * blockDim.x + threadIdx.x;
    long e = t >> 5;
    int lane = (int)(t & 31);
    if (e >= E) return;
    int s = src[e], d = dst[e];
    float coeff = dis[s] * dis[d];
    const float4 hv = *(const float4*)(h + (long)s * 128 + lane * 4);
    float* out = agg + (long)d * 128 + lane * 4;
    atomicAdd(out + 0, hv.x * coeff);
    atomicAdd(out + 1, hv.y * coeff);
    atomicAdd(out + 2, hv.z * coeff);
    atomicAdd(out + 3, hv.w * coeff);
}

// agg[i,f] += h[i,f] * dis[i]^2 + b[f]   (self-loop term + bias)
__global__ void selfbias_kernel(const float* __restrict__ h, float* agg,
                                const float* __restrict__ dis,
                                const float* __restrict__ bias, int n) {
    long t = (long)blockIdx.x * blockDim.x + threadIdx.x;
    if (t >= (long)n * 128) return;
    int i = (int)(t >> 7), f = (int)(t & 127);
    float di = dis[i];
    agg[t] += h[t] * di * di + bias[f];
}

// ---------------------------------------------------------------------------
// Host-side pipeline
// ---------------------------------------------------------------------------
extern "C" void kernel_launch(void* const* d_in, const int* in_sizes, int n_in,
                              void* d_out, int out_size, void* d_ws, size_t ws_size,
                              hipStream_t stream) {
    const float* des   = (const float*)d_in[0];
    const float* tweet = (const float*)d_in[1];
    const float* num   = (const float*)d_in[2];
    const float* cat   = (const float*)d_in[3];
    const float* nf    = (const float*)d_in[4];
    const int*   ei    = (const int*)d_in[5];
    const int E = in_sizes[6];           // edge_type length
    const int N = in_sizes[4];           // new_feature length
    const int* src = ei;
    const int* dst = ei + E;

    const float *Wd = (const float*)d_in[7],  *bd = (const float*)d_in[8];
    const float *Wt = (const float*)d_in[9],  *bt = (const float*)d_in[10];
    const float *Wn = (const float*)d_in[11], *bn = (const float*)d_in[12];
    const float *Wc = (const float*)d_in[13], *bc = (const float*)d_in[14];
    const float *Wf = (const float*)d_in[15], *bf = (const float*)d_in[16];
    const float *Wi = (const float*)d_in[17], *bi = (const float*)d_in[18];
    const float *W1 = (const float*)d_in[19], *b1 = (const float*)d_in[20];
    const float *W2 = (const float*)d_in[21], *b2 = (const float*)d_in[22];
    const float *Wo1 = (const float*)d_in[23], *bo1 = (const float*)d_in[24];
    const float *Wo2 = (const float*)d_in[25], *bo2 = (const float*)d_in[26];

    // ---- workspace layout (f32 activations + f16 packed weights) ----------
    float* xb  = (float*)d_ws;                 // [N,128]
    float* yb  = xb + (size_t)N * 128;         // [N,128]
    float* dis = yb + (size_t)N * 128;         // [N]
    size_t off = ((size_t)N * 256 + N + 15) & ~(size_t)15;   // 64B align
    v16h* pWd  = (v16h*)((float*)d_ws + off);  // 2 *24*32 frags
    v16h* pWt  = pWd + 2 * 24 * 32;            // 3 *24*32 frags
    v16h* pWi  = pWt + 3 * 24 * 32;            // 8 * 4*32 frags
    v16h* pW1  = pWi + 8 * 4 * 32;
    v16h* pW2  = pW1 + 8 * 4 * 32;
    v16h* pWo1 = pW2 + 8 * 4 * 32;
    v16h* pWo2 = pWo1 + 8 * 4 * 32;            // 1 * 4*32 frags

    const int tiles    = (N + 15) / 16;
    const int gemmGrid = (tiles + 7) / 8;      // 8 waves / 256-thread block
    const dim3 blk(256);
    const int gN    = (N + 255) / 256;
    const int gE    = (E + 255) / 256;
    const int gNF   = (int)(((size_t)N * 128 + 255) / 256);
    const int gScat = (int)(((size_t)E * 32 + 255) / 256);

    // 0) pre-pack all WMMA weights (tiny, L2-resident afterwards)
    pack_b_kernel<<<(2 * 24 * 32 + 255) / 256, blk, 0, stream>>>(Wd,  pWd, 24,  28, 2);
    pack_b_kernel<<<(3 * 24 * 32 + 255) / 256, blk, 0, stream>>>(Wt,  pWt, 24,  36, 3);
    pack_b_kernel<<<(8 * 4 * 32 + 255) / 256, blk, 0, stream>>>(Wi,  pWi,  4, 128, 8);
    pack_b_kernel<<<(8 * 4 * 32 + 255) / 256, blk, 0, stream>>>(W1,  pW1,  4, 128, 8);
    pack_b_kernel<<<(8 * 4 * 32 + 255) / 256, blk, 0, stream>>>(W2,  pW2,  4, 128, 8);
    pack_b_kernel<<<(8 * 4 * 32 + 255) / 256, blk, 0, stream>>>(Wo1, pWo1, 4, 128, 8);
    pack_b_kernel<<<(1 * 4 * 32 + 255) / 256, blk, 0, stream>>>(Wo2, pWo2, 4,   2, 1);

    // 1) x = lrelu(concat(encoders))                                  [N,128]
    encoder_kernel<<<gemmGrid, blk, 0, stream>>>(
        des, tweet, num, cat, nf, pWd, bd, pWt, bt, Wn, bn, Wc, bc, Wf, bf, xb, N);

    // 2) x = lrelu(x @ W_in + b_in)   (in-place)
    gemm128_kernel<128, true, true><<<gemmGrid, blk, 0, stream>>>(xb, pWi, bi, xb, N);

    // 3) dis = rsqrt(deg), deg = indegree + self-loop
    fill_kernel<<<gN, blk, 0, stream>>>(dis, 1.0f, N);
    deg_kernel<<<gE, blk, 0, stream>>>(dis, dst, E);
    rsqrt_kernel<<<gN, blk, 0, stream>>>(dis, N);

    // 4) GCN layer 1: h = x@W1 (in-place), yb = scatter(h) + self + b1
    gemm128_kernel<128, false, false><<<gemmGrid, blk, 0, stream>>>(xb, pW1, nullptr, xb, N);
    fill_kernel<<<gNF, blk, 0, stream>>>(yb, 0.0f, (long)N * 128);
    scatter_kernel<<<gScat, blk, 0, stream>>>(xb, yb, dis, src, dst, E);
    selfbias_kernel<<<gNF, blk, 0, stream>>>(xb, yb, dis, b1, N);

    // 5) GCN layer 2: h = yb@W2 (in-place), xb = scatter(h) + self + b2
    gemm128_kernel<128, false, false><<<gemmGrid, blk, 0, stream>>>(yb, pW2, nullptr, yb, N);
    fill_kernel<<<gNF, blk, 0, stream>>>(xb, 0.0f, (long)N * 128);
    scatter_kernel<<<gScat, blk, 0, stream>>>(yb, xb, dis, src, dst, E);
    selfbias_kernel<<<gNF, blk, 0, stream>>>(yb, xb, dis, b2, N);

    // 6) x = lrelu(x @ W_o1 + b_o1)   (in-place)
    gemm128_kernel<128, true, true><<<gemmGrid, blk, 0, stream>>>(xb, pWo1, bo1, xb, N);

    // 7) out = x @ W_o2 + b_o2                                        [N,2]
    gemm128_kernel<2, false, true><<<gemmGrid, blk, 0, stream>>>(xb, pWo2, bo2,
                                                                 (float*)d_out, N);
}
